// graphcl_30588757082541
// MI455X (gfx1250) — matrix operand order repaired and checked by reading, not compiled, not run on hardware
//
#include <hip/hip_runtime.h>
#include <cstdint>

#define D 128
#define LSTRIDE 136   // 128 + 8 bf16 pad -> 272B row stride, breaks LDS bank conflicts, keeps 16B align

typedef __attribute__((ext_vector_type(16))) __bf16        v16bf;
typedef __attribute__((ext_vector_type(8)))  float         v8f;
typedef __attribute__((ext_vector_type(4)))  unsigned int  v4u;
typedef __attribute__((ext_vector_type(4)))  int           v4i;
typedef __attribute__((ext_vector_type(8)))  int           v8i;

#if defined(__has_builtin)
#  if __has_builtin(__builtin_amdgcn_tensor_load_to_lds) && __has_builtin(__builtin_amdgcn_s_wait_tensorcnt)
#    define HAVE_TDM 1
#  endif
#endif
#ifndef HAVE_TDM
#  define HAVE_TDM 0
#endif

__device__ __forceinline__ float4 f4add(float4 a, float4 b) {
    return make_float4(a.x + b.x, a.y + b.y, a.z + b.z, a.w + b.w);
}

// B fragment (32x16 bf16): lane L holds column n = L%16; e=0..7 -> k = (L/16)*8 + e,
// e=8..15 -> k = 16 + (L/16)*8 + (e-8). With W stored transposed in LDS (Wt[n][k]), each lane reads 2x ds_load_b128.
__device__ __forceinline__ v16bf load_bfrag(const __bf16* sW, int j, int kk, int lane) {
    int n  = j * 16 + (lane & 15);
    int kb = kk * 32 + (lane >> 4) * 8;
    const uint4* q = (const uint4*)(sW + n * LSTRIDE + kb);
    union { uint4 u[2]; v16bf v; } un;
    un.u[0] = q[0];
    un.u[1] = q[2];
    return un.v;
}

// A fragment (16x32 bf16): lane L holds row m = L%16; elements e=0..7 -> k = (L/16)*8 + e,
// elements e=8..15 -> k = 16 + (L/16)*8 + (e-8). Two contiguous 8-float runs per k-step.
__device__ __forceinline__ v16bf pack_a(float4 f0, float4 f1, float4 f2, float4 f3) {
    v16bf a;
    a[0]  = (__bf16)f0.x; a[1]  = (__bf16)f0.y; a[2]  = (__bf16)f0.z; a[3]  = (__bf16)f0.w;
    a[4]  = (__bf16)f1.x; a[5]  = (__bf16)f1.y; a[6]  = (__bf16)f1.z; a[7]  = (__bf16)f1.w;
    a[8]  = (__bf16)f2.x; a[9]  = (__bf16)f2.y; a[10] = (__bf16)f2.z; a[11] = (__bf16)f2.w;
    a[12] = (__bf16)f3.x; a[13] = (__bf16)f3.y; a[14] = (__bf16)f3.z; a[15] = (__bf16)f3.w;
    return a;
}

// Pull the pre-transposed bf16 weight image (D*LSTRIDE elems = 34816 B) from global into
// LDS via the Tensor Data Mover: one TENSOR_LOAD_TO_LDS issued by lane 0, tracked by
// TENSORcnt, then a workgroup barrier. Falls back to a plain copy loop without TDM support.
__device__ __forceinline__ void tdm_stage(const __bf16* __restrict__ gsrc, __bf16* sW) {
#if HAVE_TDM
    if (threadIdx.x == 0) {
        uint32_t lds_off = (uint32_t)(uintptr_t)sW;          // generic LDS addr: low 32 bits = offset
        uint64_t ga      = (uint64_t)(uintptr_t)gsrc;
        const uint32_t nelem = (D * LSTRIDE * 2) / 4;        // 8704 f32-sized elements
        // D# group 0: count=1 | lds_addr | global_addr[56:0] | type=2 ("image")
        v4u g0 = { 1u,
                   lds_off,
                   (uint32_t)(ga & 0xffffffffu),
                   (uint32_t)((ga >> 32) & 0x01ffffffu) | (2u << 30) };
        // D# group 1: wg_mask=0, data_size=2 (4B); tensor_dim0=nelem, tensor_dim1=1,
        // tile_dim0=nelem (1-D tile), tile_dim1/2=0, dim0_stride=nelem
        v8i g1 = { (int)(2u << 16),
                   (int)((nelem & 0xffffu) << 16),
                   (int)(((nelem >> 16) & 0xffffu) | (1u << 16)),
                   (int)((nelem & 0xffffu) << 16),
                   0,
                   (int)nelem,
                   0, 0 };
        v4i z4 = { 0, 0, 0, 0 };
#if __clang_major__ >= 23
        v8i z8 = { 0, 0, 0, 0, 0, 0, 0, 0 };
        __builtin_amdgcn_tensor_load_to_lds(g0, g1, z4, z4, z8, 0);
#else
        __builtin_amdgcn_tensor_load_to_lds(g0, g1, z4, z4, 0);
#endif
        __builtin_amdgcn_s_wait_tensorcnt(0);
    }
    __syncthreads();
#else
    for (int t = threadIdx.x; t < D * LSTRIDE; t += blockDim.x) sW[t] = gsrc[t];
    __syncthreads();
#endif
}

// One-time prep: build the padded, transposed bf16 LDS image of W (row-major [k][n]) in global ws.
__global__ void prep_weights(const float* __restrict__ W, __bf16* __restrict__ wp, int total) {
    int t = blockIdx.x * blockDim.x + threadIdx.x;
    if (t >= total) return;
    int n    = t / LSTRIDE;
    int kpos = t - n * LSTRIDE;
    wp[t] = (kpos < D) ? (__bf16)W[kpos * D + n] : (__bf16)0.0f;
}

// scores[i] = sigmoid(relu(x[i]@W1 + b1) @ W2 + b2). One wave = 16 rows, WMMA over K=128.
__global__ void __launch_bounds__(256) score_kernel(
    const float* __restrict__ x, const __bf16* __restrict__ w1p,
    const float* __restrict__ b1, const float* __restrict__ W2,
    const float* __restrict__ b2, float* __restrict__ keys, int N)
{
    __shared__ __align__(16) __bf16 sW[D * LSTRIDE];
    tdm_stage(w1p, sW);

    int lane = threadIdx.x & 31;
    int wv   = threadIdx.x >> 5;
    int row0 = (blockIdx.x * 8 + wv) * 16;
    if (row0 >= N) return;

    int hh = lane >> 4;
    int rl = lane & 15;
    int row = row0 + rl;
    if (row >= N) row = N - 1;
    const float* xr = x + (size_t)row * D;
    __builtin_prefetch(xr + 128 * D, 0, 1);   // global_prefetch_b8 for next block's rows

    v16bf a[4];
#pragma unroll
    for (int kk = 0; kk < 4; ++kk) {
        int kb = kk * 32 + hh * 8;
        const float4* p  = (const float4*)(xr + kb);
        const float4* p2 = (const float4*)(xr + kb + 16);
        a[kk] = pack_a(p[0], p[1], p2[0], p2[1]);
    }

    float sp[8] = {0.f, 0.f, 0.f, 0.f, 0.f, 0.f, 0.f, 0.f};
#pragma unroll
    for (int j = 0; j < 8; ++j) {
        v8f acc = {0.f, 0.f, 0.f, 0.f, 0.f, 0.f, 0.f, 0.f};
#pragma unroll
        for (int kk = 0; kk < 4; ++kk) {
            v16bf b = load_bfrag(sW, j, kk, lane);
            acc = __builtin_amdgcn_wmma_f32_16x16x32_bf16(
                false, a[kk], false, b, (short)0, acc, false, false);
        }
        int   n   = j * 16 + rl;        // this lane's output column
        float bn  = b1[n];
        float w2n = W2[n];
#pragma unroll
        for (int r = 0; r < 8; ++r) {   // d[r] = hidden[hh*8+r][n]
            float hv = acc[r] + bn;
            hv = fmaxf(hv, 0.f);
            sp[r] += hv * w2n;
        }
    }
    // reduce partial dot-products across the 16 lanes sharing each row half
#pragma unroll
    for (int r = 0; r < 8; ++r) {
#pragma unroll
        for (int m = 8; m >= 1; m >>= 1) sp[r] += __shfl_xor(sp[r], m, 16);
    }
    if (rl == 0) {
        float bb = b2[0];
#pragma unroll
        for (int r = 0; r < 8; ++r) {
            int orow = row0 + hh * 8 + r;
            if (orow < N) keys[orow] = 1.f / (1.f + __expf(-(sp[r] + bb)));
        }
    }
}

// out = relu((xin + agg) @ Wg + bg)
__global__ void __launch_bounds__(256) node_update_kernel(
    const float* __restrict__ xin, const float* __restrict__ agg,
    const __bf16* __restrict__ wgp, const float* __restrict__ bg,
    float* __restrict__ out, int N)
{
    __shared__ __align__(16) __bf16 sW[D * LSTRIDE];
    tdm_stage(wgp, sW);

    int lane = threadIdx.x & 31;
    int wv   = threadIdx.x >> 5;
    int row0 = (blockIdx.x * 8 + wv) * 16;
    if (row0 >= N) return;

    int hh = lane >> 4;
    int rl = lane & 15;
    int row = row0 + rl;
    if (row >= N) row = N - 1;
    const float* xr = xin + (size_t)row * D;
    const float* ar = agg + (size_t)row * D;

    v16bf a[4];
#pragma unroll
    for (int kk = 0; kk < 4; ++kk) {
        int kb = kk * 32 + hh * 8;
        const float4* px  = (const float4*)(xr + kb);
        const float4* pa  = (const float4*)(ar + kb);
        const float4* px2 = (const float4*)(xr + kb + 16);
        const float4* pa2 = (const float4*)(ar + kb + 16);
        a[kk] = pack_a(f4add(px[0], pa[0]), f4add(px[1], pa[1]),
                       f4add(px2[0], pa2[0]), f4add(px2[1], pa2[1]));
    }

#pragma unroll
    for (int j = 0; j < 8; ++j) {
        v8f acc = {0.f, 0.f, 0.f, 0.f, 0.f, 0.f, 0.f, 0.f};
#pragma unroll
        for (int kk = 0; kk < 4; ++kk) {
            v16bf b = load_bfrag(sW, j, kk, lane);
            acc = __builtin_amdgcn_wmma_f32_16x16x32_bf16(
                false, a[kk], false, b, (short)0, acc, false, false);
        }
        int   n  = j * 16 + rl;
        float bn = bg[n];
#pragma unroll
        for (int r = 0; r < 8; ++r) {
            int orow = row0 + hh * 8 + r;
            if (orow < N) out[(size_t)orow * D + n] = fmaxf(acc[r] + bn, 0.f);
        }
    }
}

// One wave per edge: msg = xin[src] + sum(attr_emb[ai]) + type_emb[ti]; agg[dst] += msg
__global__ void __launch_bounds__(256) edge_agg(
    const float* __restrict__ xin, const int* __restrict__ eidx,
    const float* __restrict__ eattr, const float* __restrict__ etype,
    const float* __restrict__ attr_emb, const float* __restrict__ type_emb,
    float* __restrict__ agg, int E, int N, int VA, int VT)
{
    int e = blockIdx.x * (blockDim.x >> 5) + (threadIdx.x >> 5);
    if (e >= E) return;
    int lane = threadIdx.x & 31;

    int src = eidx[e];
    int dst = eidx[E + e];
    src = (src < 0) ? 0 : ((src >= N) ? N - 1 : src);
    dst = (dst < 0) ? 0 : ((dst >= N) ? N - 1 : dst);

    int a0 = (int)eattr[e * 4 + 0]; a0 = a0 < 0 ? 0 : (a0 >= VA ? VA - 1 : a0);
    int a1 = (int)eattr[e * 4 + 1]; a1 = a1 < 0 ? 0 : (a1 >= VA ? VA - 1 : a1);
    int a2 = (int)eattr[e * 4 + 2]; a2 = a2 < 0 ? 0 : (a2 >= VA ? VA - 1 : a2);
    int a3 = (int)eattr[e * 4 + 3]; a3 = a3 < 0 ? 0 : (a3 >= VA ? VA - 1 : a3);
    int t0 = (int)etype[e];         t0 = t0 < 0 ? 0 : (t0 >= VT ? VT - 1 : t0);

    int c = lane * 4;
    float4 v = *(const float4*)(xin + (size_t)src * D + c);
    v = f4add(v, *(const float4*)(attr_emb + (size_t)a0 * D + c));
    v = f4add(v, *(const float4*)(attr_emb + (size_t)a1 * D + c));
    v = f4add(v, *(const float4*)(attr_emb + (size_t)a2 * D + c));
    v = f4add(v, *(const float4*)(attr_emb + (size_t)a3 * D + c));
    v = f4add(v, *(const float4*)(type_emb + (size_t)t0 * D + c));

    float* ap = agg + (size_t)dst * D + c;
    atomicAdd(ap + 0, v.x);
    atomicAdd(ap + 1, v.y);
    atomicAdd(ap + 2, v.z);
    atomicAdd(ap + 3, v.w);
}

__global__ void fill_f32(float* p, float v, int n) {
    int i = blockIdx.x * blockDim.x + threadIdx.x;
    if (i < n) p[i] = v;
}
__global__ void fill_i32(int* p, int v, int n) {
    int i = blockIdx.x * blockDim.x + threadIdx.x;
    if (i < n) p[i] = v;
}
__global__ void copy_f32(const float* __restrict__ a, float* __restrict__ b, int n) {
    int i = blockIdx.x * blockDim.x + threadIdx.x;
    if (i < n) b[i] = a[i];
}
__global__ void sort_init(float* key, int* val, int N, int M) {
    int i = blockIdx.x * blockDim.x + threadIdx.x;
    if (i >= M) return;
    val[i] = i;
    if (i >= N) key[i] = -3.0e38f;
}
// Descending bitonic step
__global__ void bitonic_step(float* key, int* val, int j, int size, int M) {
    int i = blockIdx.x * blockDim.x + threadIdx.x;
    if (i >= M) return;
    int ixj = i ^ j;
    if (ixj > i) {
        float ki = key[i], kj = key[ixj];
        bool desc = ((i & size) == 0);
        bool sw = desc ? (ki < kj) : (ki > kj);
        if (sw) {
            key[i] = kj; key[ixj] = ki;
            int vi = val[i]; val[i] = val[ixj]; val[ixj] = vi;
        }
    }
}
__global__ void mark_topk(const int* __restrict__ val, int* __restrict__ flags,
                          int* __restrict__ idx_out, int k, int N) {
    int t = blockIdx.x * blockDim.x + threadIdx.x;
    if (t >= k) return;
    int v = val[t];
    idx_out[t] = v;
    if (v >= 0 && v < N) flags[v] = 1;
}
__global__ void build_xm(const float* __restrict__ x, const int* __restrict__ flags,
                         const float* __restrict__ mask_emb, float* __restrict__ xm, int ND) {
    int i = blockIdx.x * blockDim.x + threadIdx.x;
    if (i >= ND) return;
    int c = i & 127;
    int n = i >> 7;
    xm[i] = flags[n] ? mask_emb[c] : x[i];
}
__global__ void mask_edges(const int* __restrict__ nindex, const float* __restrict__ mask_attr,
                           const float* __restrict__ mask_type, float* __restrict__ ea,
                           float* __restrict__ et, int ne, int E) {
    int t = blockIdx.x * blockDim.x + threadIdx.x;
    if (t >= ne) return;
    int e = nindex[t];
    if (e < 0 || e >= E) return;
    ea[e * 4 + 0] = mask_attr[0];
    ea[e * 4 + 1] = mask_attr[1];
    ea[e * 4 + 2] = mask_attr[2];
    ea[e * 4 + 3] = mask_attr[3];
    et[e] = mask_type[0];
}

extern "C" void kernel_launch(void* const* d_in, const int* in_sizes, int n_in,
                              void* d_out, int out_size, void* d_ws, size_t ws_size,
                              hipStream_t stream) {
    const float* x         = (const float*)d_in[0];
    const int*   eidx      = (const int*)  d_in[1];
    const float* eattr     = (const float*)d_in[2];
    const float* etype     = (const float*)d_in[3];
    const int*   nindex    = (const int*)  d_in[4];
    // d_in[5] = k (device scalar; recomputed on host from sizes instead)
    const float* W1        = (const float*)d_in[6];
    const float* b1        = (const float*)d_in[7];
    const float* W2        = (const float*)d_in[8];
    const float* b2        = (const float*)d_in[9];
    const float* mask_emb  = (const float*)d_in[10];
    const float* mask_attr = (const float*)d_in[11];
    const float* mask_type = (const float*)d_in[12];
    const float* attr_emb  = (const float*)d_in[13];
    const float* type_emb  = (const float*)d_in[14];
    const float* Wg        = (const float*)d_in[15];
    const float* bg        = (const float*)d_in[16];

    const int ND = in_sizes[0];
    const int N  = ND / D;
    const int E  = in_sizes[2] / 4;
    const int NE = in_sizes[4];
    const int VA = in_sizes[13] / D;
    const int VT = in_sizes[14] / D;
    const int k  = out_size - 2 * ND - 5 * E;   // hnew + h + idx + ea(E*4) + et(E)

    // output layout (flat, return order): hnew[N,D], h[N,D], idx[k], ea[E,4], et[E,1]
    float* out  = (float*)d_out;
    float* hnew = out;
    float* hbuf = out + (size_t)ND;
    int*   idxo = (int*)d_out + 2 * (size_t)ND;
    float* ea   = out + 2 * (size_t)ND + (size_t)k;
    float* et   = ea + (size_t)E * 4;

    int M = 1;
    while (M < N) M <<= 1;   // 65536 for N=50000

    // workspace carve-out
    size_t off = 0;
    auto take = [&](size_t bytes) -> void* {
        void* p = (char*)d_ws + off;
        off = (off + bytes + 255) & ~(size_t)255;
        return p;
    };
    const int WPE = D * LSTRIDE;                    // prepped weight image elements
    float*  keys  = (float*) take((size_t)M * 4);
    int*    vals  = (int*)   take((size_t)M * 4);
    int*    flags = (int*)   take((size_t)N * 4);
    float*  xm    = (float*) take((size_t)ND * 4);
    float*  agg   = (float*) take((size_t)ND * 4);
    __bf16* wp1   = (__bf16*)take((size_t)WPE * 2);
    __bf16* wpg   = (__bf16*)take((size_t)WPE * 2);
    (void)ws_size; (void)n_in;

    dim3 b256(256);
    int gemmBlocks = (N + 127) / 128;   // 128 rows per block (8 waves x 16 rows)

    // 0. one-time weight prep (transposed, padded, bf16) for TDM staging
    prep_weights<<<(WPE + 255) / 256, b256, 0, stream>>>(W1, wp1, WPE);
    prep_weights<<<(WPE + 255) / 256, b256, 0, stream>>>(Wg, wpg, WPE);

    // 1. scoring MLP (WMMA, TDM-staged weights)
    score_kernel<<<gemmBlocks, b256, 0, stream>>>(x, wp1, b1, W2, b2, keys, N);

    // 2. top-k via full descending bitonic sort over M=2^ceil(log2 N)
    sort_init<<<(M + 255) / 256, b256, 0, stream>>>(keys, vals, N, M);
    for (int size = 2; size <= M; size <<= 1)
        for (int j = size >> 1; j > 0; j >>= 1)
            bitonic_step<<<(M + 255) / 256, b256, 0, stream>>>(keys, vals, j, size, M);
    fill_i32<<<(N + 255) / 256, b256, 0, stream>>>(flags, 0, N);
    mark_topk<<<(k + 255) / 256, b256, 0, stream>>>(vals, flags, idxo, k, N);

    // 3. mask node features
    build_xm<<<(ND + 255) / 256, b256, 0, stream>>>(x, flags, mask_emb, xm, ND);

    // 4. GNN layer 1: scatter-aggregate then fused (xm+agg)@Wg WMMA GEMM -> h
    fill_f32<<<(ND + 255) / 256, b256, 0, stream>>>(agg, 0.f, ND);
    edge_agg<<<(E + 7) / 8, b256, 0, stream>>>(xm, eidx, eattr, etype,
                                               attr_emb, type_emb, agg, E, N, VA, VT);
    node_update_kernel<<<gemmBlocks, b256, 0, stream>>>(xm, agg, wpg, bg, hbuf, N);

    // 5. edge feature masking -> ea, et (outputs)
    copy_f32<<<((E * 4) + 255) / 256, b256, 0, stream>>>(eattr, ea, E * 4);
    copy_f32<<<(E + 255) / 256, b256, 0, stream>>>(etype, et, E);
    mask_edges<<<(NE + 255) / 256, b256, 0, stream>>>(nindex, mask_attr, mask_type, ea, et, NE, E);

    // 6. GNN layer 2 on h with masked edge features -> hnew
    fill_f32<<<(ND + 255) / 256, b256, 0, stream>>>(agg, 0.f, ND);
    edge_agg<<<(E + 7) / 8, b256, 0, stream>>>(hbuf, eidx, ea, et,
                                               attr_emb, type_emb, agg, E, N, VA, VT);
    node_update_kernel<<<gemmBlocks, b256, 0, stream>>>(hbuf, agg, wpg, bg, hnew, N);
}
